// KeystrokeAttention_58265526337975
// MI455X (gfx1250) — matrix operand order repaired and checked by reading, not compiled, not run on hardware
//
#include <hip/hip_runtime.h>
#include <math.h>

// ---------------------------------------------------------------------------
// Additive attention, MI455X (gfx1250, wave32, WMMA), single-pass fused.
//
//   energy[b,s] = enc[b,s]·W_e   (hidden·W_h + bias drops out of softmax)
//   attn        = softmax_s(energy)
//   context[b,h]= sum_s attn[b,s] * enc[b,s,h]
//
// enc = 256 MB f32, everything else KB-scale -> pure HBM roofline.
//   two-pass  : 512 MB  ~22 us @ 23.3 TB/s
//   fused     : 256 MB + 16 MB partial traffic ~11.7 us  <-- this file
//
// Fused kernel (one block = one 32-row s-chunk of one batch):
//   1. Stream 32x1024 f32 slab to LDS; fold the enc·W_e dot into the fill
//      (f32 register accumulation, 8 threads/row, shuffle-reduced).
//   2. Chunk-local softmax stats m_i, l_i; A-operand = exp(e-m_i) (f16).
//   3. Context: per wave 8 h-tiles, one v_wmma_f32_16x16x32_f16 each
//      (K=32 = whole chunk), B gathered from LDS.
//   4. Combine kernel merges the 64 chunk-partials per batch.
// ---------------------------------------------------------------------------

typedef __attribute__((ext_vector_type(16))) _Float16 v16h;
typedef __attribute__((ext_vector_type(8)))  float    v8f;

#define BATCH 32
#define SLEN  2048
#define HDIM  1024
#define CHUNK 32                    // s-rows per fused block
#define NCHUNK (SLEN / CHUNK)       // 64 chunks per batch
#define LDSTRIDE 1028               // floats per LDS row (pad 4 -> vary banks)

// Pack two float4 (8 consecutive f32) into v16h[base..base+7] as f16.
__device__ __forceinline__ void pack8(v16h& v, int base, const float4& x, const float4& y) {
    v[base + 0] = (_Float16)x.x; v[base + 1] = (_Float16)x.y;
    v[base + 2] = (_Float16)x.z; v[base + 3] = (_Float16)x.w;
    v[base + 4] = (_Float16)y.x; v[base + 5] = (_Float16)y.y;
    v[base + 6] = (_Float16)y.z; v[base + 7] = (_Float16)y.w;
}

// ===========================================================================
// FUSED single-pass kernel: 2048 blocks x 256 threads.
// ===========================================================================
__global__ void attn_fused_chunk(const float* __restrict__ enc,
                                 const float* __restrict__ W,
                                 float* __restrict__ pctx,   // [2048][1024]
                                 float* __restrict__ m_arr,  // [2048]
                                 float* __restrict__ l_arr)  // [2048]
{
    __shared__ float slab[CHUNK * LDSTRIDE];   // 128.5 KB f32 slab
    __shared__ float eSh[CHUNK];

    const int chunk = blockIdx.x;              // 0..2047
    const int b     = chunk >> 6;              // /NCHUNK
    const int s0    = (chunk & (NCHUNK - 1)) * CHUNK;
    const int t     = threadIdx.x;             // 0..255
    const int lane  = t & 31;
    const int wv    = t >> 5;                  // wave 0..7
    const int half  = lane >> 4;
    const int n     = lane & 15;

    // ---- Phase 1: fill LDS slab + fused enc·W_e dot -----------------------
    // thread t -> row r = t>>3, owns float4 columns u + 8*i  (u = t&7)
    {
        const int r = t >> 3;
        const int u = t & 7;
        const float4* grow = reinterpret_cast<const float4*>(
            enc + ((size_t)b * SLEN + s0 + r) * HDIM);
        const float4* wq = reinterpret_cast<const float4*>(W + HDIM); // W_e
        float4* lrow = reinterpret_cast<float4*>(slab + r * LDSTRIDE);

        float part = 0.f;
        #pragma unroll 8
        for (int i = 0; i < HDIM / 4 / 8; ++i) {     // 32 float4 per thread
            const int j = u + 8 * i;
            __builtin_prefetch(grow + j + 64, 0, 0); // global_prefetch_b8
            float4 v = grow[j];
            float4 w4 = wq[j];
            part = fmaf(v.x, w4.x, part);
            part = fmaf(v.y, w4.y, part);
            part = fmaf(v.z, w4.z, part);
            part = fmaf(v.w, w4.w, part);
            lrow[j] = v;
        }
        // reduce the 8 column-partials of each row (lanes t&7 grouped)
        part += __shfl_xor(part, 4, 32);
        part += __shfl_xor(part, 2, 32);
        part += __shfl_xor(part, 1, 32);
        if (u == 0) eSh[r] = part;                   // energy of row r
    }
    __syncthreads();

    // ---- Phase 2: chunk softmax stats + A operand -------------------------
    float m = -INFINITY;
    #pragma unroll
    for (int s = 0; s < CHUNK; ++s) m = fmaxf(m, eSh[s]);   // LDS broadcast
    float l = 0.f;
    #pragma unroll
    for (int s = 0; s < CHUNK; ++s) l += __expf(eSh[s] - m);
    if (t == 0) { m_arr[chunk] = m; l_arr[chunk] = l; }

    // A layout (16-bit 16x32): lane -> K run [16*half, +16);
    // rows replicated => every D row equals the context tile.
    v16h a;
    #pragma unroll
    for (int j = 0; j < 8; ++j) {
        a[j]     = (_Float16)__expf(eSh[half * 16 + j]      - m);
        a[8 + j] = (_Float16)__expf(eSh[half * 16 + 8 + j] - m);
    }

    // ---- Phase 3: context partial via WMMA, 8 h-tiles per wave ------------
    // B layout (32x16): lane -> col n, K run [16*half, +16).
    #pragma unroll
    for (int tile = 0; tile < 8; ++tile) {
        const int h0 = wv * 128 + tile * 16;
        const float* col = slab + (half * 16) * LDSTRIDE + h0 + n;
        v16h bv;
        #pragma unroll
        for (int e = 0; e < 16; ++e)
            bv[e] = (_Float16)col[e * LDSTRIDE];

        v8f acc = {};
        acc = __builtin_amdgcn_wmma_f32_16x16x32_f16(
            false, a, false, bv, (short)0, acc, false, false);

        if (half == 0)
            pctx[((size_t)chunk << 10) + h0 + n] = acc[0];
    }
}

// ===========================================================================
// Combine kernel: one block per batch, merge 64 chunk partials.
//   ctx[h] = sum_i e^{m_i-M} ctx_i[h] / sum_i e^{m_i-M} l_i
// ===========================================================================
__global__ void attn_combine(const float* __restrict__ pctx,
                             const float* __restrict__ m_arr,
                             const float* __restrict__ l_arr,
                             float* __restrict__ out)
{
    __shared__ float wSh[NCHUNK];
    const int b = blockIdx.x;
    const int t = threadIdx.x;

    float M = -INFINITY;
    #pragma unroll 8
    for (int i = 0; i < NCHUNK; ++i) M = fmaxf(M, m_arr[b * NCHUNK + i]);
    float denom = 0.f;
    #pragma unroll 8
    for (int i = 0; i < NCHUNK; ++i)
        denom += __expf(m_arr[b * NCHUNK + i] - M) * l_arr[b * NCHUNK + i];
    if (t < NCHUNK) wSh[t] = __expf(m_arr[b * NCHUNK + t] - M);
    __syncthreads();

    const float inv = 1.f / denom;
    #pragma unroll
    for (int hb = 0; hb < HDIM / 256; ++hb) {
        const int h = t + hb * 256;
        float acc = 0.f;
        #pragma unroll 8
        for (int i = 0; i < NCHUNK; ++i)
            acc += wSh[i] * pctx[(((size_t)b * NCHUNK + i) << 10) + h];
        out[(size_t)b * HDIM + h] = acc * inv;
    }
}

// ===========================================================================
// Fallback two-pass pipeline (used when workspace < 8 MB): round-1
// WMMA kernels. 512 MB traffic (~22 us) instead of ~272 MB.
// ===========================================================================
__global__ void attn_energy_wmma(const float* __restrict__ enc,
                                 const float* __restrict__ W,
                                 float* __restrict__ energy) {
    const int lane  = threadIdx.x & 31;
    const int wave  = (blockIdx.x * blockDim.x + threadIdx.x) >> 5;
    const int b     = wave >> 7;
    const int s0    = (wave & 127) << 4;
    const int m     = lane & 15;
    const int half  = lane >> 4;

    const float* rowA = enc + ((size_t)b * SLEN + (s0 + m)) * HDIM;
    const float* We   = W + HDIM;

    v8f acc = {};
    for (int k0 = 0; k0 < HDIM; k0 += 32) {
        __builtin_prefetch(rowA + k0 + 256, 0, 0);
        const float4* pa0 = reinterpret_cast<const float4*>(rowA + k0 + half * 16);
        const float4* pa1 = reinterpret_cast<const float4*>(rowA + k0 + half * 16 + 8);
        float4 a0 = pa0[0], a1 = pa0[1];
        float4 a2 = pa1[0], a3 = pa1[1];
        v16h a;
        pack8(a, 0, a0, a1);
        pack8(a, 8, a2, a3);

        const float4* pb = reinterpret_cast<const float4*>(We + k0 + half * 16);
        float4 b0 = pb[0], b1 = pb[1], b2 = pb[2], b3 = pb[3];
        v16h bv;
        pack8(bv, 0, b0, b1);
        pack8(bv, 8, b2, b3);

        acc = __builtin_amdgcn_wmma_f32_16x16x32_f16(
            false, a, false, bv, (short)0, acc, false, false);
    }
    if (m == 0) {
        #pragma unroll
        for (int v = 0; v < 8; ++v)
            energy[(size_t)b * SLEN + s0 + half * 8 + v] = acc[v];
    }
}

__global__ void attn_softmax(const float* __restrict__ energy,
                             float* __restrict__ attn) {
    __shared__ float red[16];
    const int b = blockIdx.x;
    const int t = threadIdx.x;
    const float* e = energy + (size_t)b * SLEN;

    float vals[SLEN / 256];
    float vmax = -INFINITY;
    #pragma unroll
    for (int i = 0; i < SLEN / 256; ++i) {
        vals[i] = e[t + i * 256];
        vmax = fmaxf(vmax, vals[i]);
    }
    #pragma unroll
    for (int off = 16; off > 0; off >>= 1)
        vmax = fmaxf(vmax, __shfl_xor(vmax, off, 32));
    if ((t & 31) == 0) red[t >> 5] = vmax;
    __syncthreads();
    float bmax = red[0];
    #pragma unroll
    for (int w = 1; w < 8; ++w) bmax = fmaxf(bmax, red[w]);
    __syncthreads();

    float s = 0.f;
    #pragma unroll
    for (int i = 0; i < SLEN / 256; ++i) {
        vals[i] = __expf(vals[i] - bmax);
        s += vals[i];
    }
    #pragma unroll
    for (int off = 16; off > 0; off >>= 1)
        s += __shfl_xor(s, off, 32);
    if ((t & 31) == 0) red[8 + (t >> 5)] = s;
    __syncthreads();
    float bsum = 0.f;
    #pragma unroll
    for (int w = 0; w < 8; ++w) bsum += red[8 + w];
    const float inv = 1.f / bsum;

    #pragma unroll
    for (int i = 0; i < SLEN / 256; ++i)
        attn[(size_t)b * SLEN + t + i * 256] = vals[i] * inv;
}

__global__ void attn_context_wmma(const float* __restrict__ enc,
                                  const float* __restrict__ attn,
                                  float* __restrict__ out) {
    const int lane = threadIdx.x & 31;
    const int wave = (blockIdx.x * blockDim.x + threadIdx.x) >> 5;
    const int b    = wave >> 6;
    const int h0   = (wave & 63) << 4;
    const int n    = lane & 15;
    const int half = lane >> 4;

    const float* encB = enc + (size_t)b * SLEN * HDIM;
    const float* at   = attn + (size_t)b * SLEN;

    v8f acc = {};
    for (int s0 = 0; s0 < SLEN; s0 += 32) {
        const float4* pa0 = reinterpret_cast<const float4*>(at + s0 + half * 16);
        const float4* pa1 = reinterpret_cast<const float4*>(at + s0 + half * 16 + 8);
        float4 a0 = pa0[0], a1 = pa0[1];
        float4 a2 = pa1[0], a3 = pa1[1];
        v16h a;
        pack8(a, 0, a0, a1);
        pack8(a, 8, a2, a3);

        const float* colB = encB + (size_t)(s0 + half * 16) * HDIM + h0 + n;
        __builtin_prefetch(colB + 32 * HDIM, 0, 0);
        v16h bv;
        #pragma unroll
        for (int e = 0; e < 16; ++e)
            bv[e] = (_Float16)colB[(size_t)e * HDIM];

        acc = __builtin_amdgcn_wmma_f32_16x16x32_f16(
            false, a, false, bv, (short)0, acc, false, false);
    }
    if (half == 0)
        out[(size_t)b * HDIM + h0 + n] = acc[0];
}

// ===========================================================================
extern "C" void kernel_launch(void* const* d_in, const int* in_sizes, int n_in,
                              void* d_out, int out_size, void* d_ws, size_t ws_size,
                              hipStream_t stream) {
    // setup_inputs order: hidden (unused), encoder_outputs, W, b (unused)
    const float* enc = (const float*)d_in[1];
    const float* W   = (const float*)d_in[2];
    float* out = (float*)d_out;

    const size_t nchunks    = (size_t)BATCH * NCHUNK;                 // 2048
    const size_t need_fused = (nchunks * HDIM + 2 * nchunks) * sizeof(float);

    if (ws_size >= need_fused) {
        // -------- single-pass fused path (~272 MB traffic) --------
        float* pctx  = (float*)d_ws;                 // 2048 x 1024
        float* m_arr = pctx + nchunks * HDIM;        // 2048
        float* l_arr = m_arr + nchunks;              // 2048
        attn_fused_chunk<<<(int)nchunks, 256, 0, stream>>>(enc, W, pctx, m_arr, l_arr);
        attn_combine<<<BATCH, 256, 0, stream>>>(pctx, m_arr, l_arr, out);
    } else {
        // -------- two-pass fallback (~512 MB traffic) --------
        float* energy = (float*)d_ws;                  // 32*2048
        float* attn   = energy + (size_t)BATCH * SLEN; // 32*2048
        attn_energy_wmma<<<(BATCH * SLEN / 16) / 8, 256, 0, stream>>>(enc, W, energy);
        attn_softmax<<<BATCH, 256, 0, stream>>>(energy, attn);
        attn_context_wmma<<<(BATCH * HDIM / 16) / 8, 256, 0, stream>>>(enc, attn, out);
    }
}
